// DetSegTransformerDecoderLayer_34076270527051
// MI455X (gfx1250) — matrix operand order repaired and checked by reading, not compiled, not run on hardware
//
#include <hip/hip_runtime.h>
#include <hip/hip_bf16.h>

// ---------------------------------------------------------------------------
// Problem constants (from the reference)
// ---------------------------------------------------------------------------
#define QB   40000
#define QD   900
#define QT   40900        // QB + QD
#define DCH  128
#define NCAM 6

typedef _Float16 half8 __attribute__((ext_vector_type(8)));
typedef _Float16 v16h  __attribute__((ext_vector_type(16)));
typedef float    v8f   __attribute__((ext_vector_type(8)));
typedef unsigned int uint4v __attribute__((ext_vector_type(4)));
typedef int      int4v __attribute__((ext_vector_type(4)));
typedef int      int8v __attribute__((ext_vector_type(8)));

union AFrag { v16h v; half8 h[2]; };

// LDS byte-offset of a generic pointer that points into LDS
static __device__ inline unsigned lds_off_of(const void* p) {
  return (unsigned)(unsigned long long)(
      (__attribute__((address_space(3))) const void*)p);
}

// ---------------------------------------------------------------------------
// TDM: 1-D contiguous copy of `halves` f16 elements global -> LDS.
// Descriptor per CDNA5 ISA ch.8 (D# group0/group1), data_size=1 (2 bytes),
// tile_dim0 = tensor_dim0 = stride0 = halves, tile_dim1 = tensor_dim1 = 1.
// ---------------------------------------------------------------------------
static __device__ inline void tdm_load_1d(const _Float16* gsrc, unsigned lds_byte_off,
                                          unsigned halves) {
  unsigned long long ga = (unsigned long long)(const void*)gsrc;
  uint4v g0;
  g0.x = 1u;                                   // count=1 (valid descriptor)
  g0.y = lds_byte_off;                         // [63:32] lds_addr
  g0.z = (unsigned)(ga & 0xFFFFFFFFu);         // [95:64] global_addr lo
  g0.w = (unsigned)((ga >> 32) & 0x01FFFFFFu)  // [120:96] global_addr hi
       | (2u << 30);                           // [127:126] type=2 (image)
  int8v g1;
  g1[0] = 0x10000;                             // data_size=1 (2B), no multicast
  g1[1] = (int)((halves & 0xFFFFu) << 16);     // tensor_dim0 lo16 -> bits[63:48]
  g1[2] = (int)((halves >> 16) | (1u << 16));  // tensor_dim0 hi16 | tensor_dim1=1
  g1[3] = (int)((halves & 0xFFFFu) << 16);     // tile_dim0 -> bits[127:112]
  g1[4] = 1;                                   // tile_dim1=1, tile_dim2=0
  g1[5] = (int)halves;                         // tensor_dim0_stride lo32
  g1[6] = 0;
  g1[7] = 0;
  int4v gz = {0, 0, 0, 0};
#if defined(__clang_major__) && (__clang_major__ >= 23)
  int8v gz8 = {0, 0, 0, 0, 0, 0, 0, 0};
  __builtin_amdgcn_tensor_load_to_lds(g0, g1, gz, gz, gz8, 0);
#else
  __builtin_amdgcn_tensor_load_to_lds(g0, g1, gz, gz, 0);
#endif
}

// ---------------------------------------------------------------------------
// Generic tiled GEMM:  C[M,N] = act(A[M,K] @ B + bias) (+ Res)
//   A: f32 row-major (M x K), K multiple of 32
//   B: f32 row-major (K x N), or (N x K) if BT
//   ACT: 0=none 1=relu 2=gelu(exact) ; RES: add residual Res[M,N]
// Block: 256 threads (8 waves), tile 128x64, ping-pong double-buffered LDS
// staging (one barrier per K-step) + global_prefetch of the K+2 A tile.
// Fragments follow CDNA5 ISA 7.12.2 wave32 layouts.
// ---------------------------------------------------------------------------
template<int ACT, int BT, int RES>
__global__ __launch_bounds__(256) void gemm_f16_wmma(
    const float* __restrict__ A, const float* __restrict__ B,
    const float* __restrict__ bias, const float* __restrict__ Res,
    float* __restrict__ C, int M, int N, int K)
{
  __shared__ __align__(16) _Float16 As[2][128][32];
  __shared__ __align__(16) _Float16 Bs[2][32][64];
  const int tid  = threadIdx.x;
  const int wave = tid >> 5, lane = tid & 31;
  const int l = lane & 15, hi = lane >> 4;
  const int mb = blockIdx.x * 128, nb = blockIdx.y * 64;

  auto stage = [&](int kb, int buf) {
    { // A tile 128x32 (f32 -> f16): 2 threads/row, 16 elems each
      int row = tid >> 1, c0 = (tid & 1) * 16;
      int m = mb + row;
      const float* src = A + (size_t)m * K + kb + c0;
      for (int j = 0; j < 16; ++j)
        As[buf][row][c0 + j] = (m < M) ? (_Float16)src[j] : (_Float16)0.f;
      if (m < M && kb + 64 < K) __builtin_prefetch(src + 64, 0, 0);
    }
    { // B tile 32x64: 8 threads/row, 8 elems each
      int row = tid >> 3, c0 = (tid & 7) * 8;
      for (int j = 0; j < 8; ++j) {
        int n = nb + c0 + j;
        float v = 0.f;
        if (n < N) v = BT ? B[(size_t)n * K + kb + row]
                          : B[(size_t)(kb + row) * N + n];
        Bs[buf][row][c0 + j] = (_Float16)v;
      }
    }
  };

  v8f acc[4];
  for (int t = 0; t < 4; ++t)
    for (int r = 0; r < 8; ++r) acc[t][r] = 0.f;

  stage(0, 0);
  __syncthreads();
  for (int kb = 0; kb < K; kb += 32) {
    int cur = (kb >> 5) & 1;
    if (kb + 32 < K) stage(kb + 32, cur ^ 1);   // overlap next stage with math

    AFrag af;   // A fragment: row m = l; K chunks {hi*8, hi*8+16}
    const _Float16* ap = &As[cur][wave * 16 + l][hi * 8];
    af.h[0] = *(const half8*)ap;
    af.h[1] = *(const half8*)(ap + 16);
    for (int t = 0; t < 4; ++t) {
      AFrag bf; // B fragment: col n = l; K chunks {hi*8, hi*8+16}
      for (int i = 0; i < 16; ++i)
        bf.v[i] = Bs[cur][hi * 8 + (i < 8 ? i : i + 8)][t * 16 + l];
      acc[t] = __builtin_amdgcn_wmma_f32_16x16x32_f16(
          false, af.v, false, bf.v, (short)0, acc[t], false, false);
    }
    __syncthreads();
  }

  for (int t = 0; t < 4; ++t) {
    int n = nb + t * 16 + l;
    if (n >= N) continue;
    for (int r = 0; r < 8; ++r) {
      int m = mb + wave * 16 + r + hi * 8;  // D layout: M = r + 8*(lane>=16)
      if (m >= M) continue;
      float x = acc[t][r] + bias[n];
      if (ACT == 1) x = fmaxf(x, 0.f);
      else if (ACT == 2) x = 0.5f * x * (1.f + erff(x * 0.70710678118654752f));
      if (RES) x += Res[(size_t)m * N + n];
      C[(size_t)m * N + n] = x;
    }
  }
}

// ---------------------------------------------------------------------------
// PE MLP: hid = relu(pos @ w1 + b1) [256], out = base + hid @ w2 + b2 [128]
// ---------------------------------------------------------------------------
__global__ __launch_bounds__(256) void pe_kernel(
    const float* __restrict__ bev_query, const float* __restrict__ bev_pos,
    const float* __restrict__ det_query, const float* __restrict__ det_pos,
    const float* __restrict__ w1, const float* __restrict__ b1,
    const float* __restrict__ w2, const float* __restrict__ b2,
    float* __restrict__ xin, float* __restrict__ det_qe)
{
  int q = blockIdx.x;
  const float* pos; const float* base; float* out;
  if (q < QB) { pos = bev_pos + (size_t)q * 3; base = bev_query + (size_t)q * DCH; out = xin + (size_t)q * DCH; }
  else { int j = q - QB; pos = det_pos + (size_t)j * 3; base = det_query + (size_t)j * DCH; out = det_qe + (size_t)j * DCH; }
  __shared__ float hid[256];
  int t = threadIdx.x;
  float p0 = pos[0], p1 = pos[1], p2 = pos[2];
  float h = p0 * w1[t] + p1 * w1[256 + t] + p2 * w1[512 + t] + b1[t];
  hid[t] = fmaxf(h, 0.f);
  __syncthreads();
  if (t < DCH) {
    float s = b2[t];
    for (int k = 0; k < 256; ++k) s += hid[k] * w2[k * DCH + t];
    out[t] = base[t] + s;
  }
}

// f32 -> f16 elementwise
__global__ void cvt_f16_kernel(const float* __restrict__ a, _Float16* __restrict__ b, int n)
{
  int i = blockIdx.x * blockDim.x + threadIdx.x;
  if (i < n) b[i] = (_Float16)a[i];
}

// inconv_w2 (O,C,5,5) f32 -> w2r[tap][c][o] f16
__global__ void w2prep_kernel(const float* __restrict__ w2, _Float16* __restrict__ wr)
{
  int i = blockIdx.x * blockDim.x + threadIdx.x;
  if (i >= 25 * 128 * 128) return;
  int o = i & 127, c = (i >> 7) & 127, tap = i >> 14;
  wr[i] = (_Float16)w2[(size_t)(o * 128 + c) * 25 + tap];
}

// ---------------------------------------------------------------------------
// 5x5 SAME conv, 128->128, implicit GEMM.  Block = 128 thr (4 waves) =
// 16 pixels (fixed row h) x 128 output channels.  Per-tap 128x128 f16
// weight slab is DMA'd into LDS by the Tensor Data Mover (one tensor_load
// per wave for its 32-row chunk), double-buffered across taps and fenced
// with s_wait_tensorcnt; B fragments then come from LDS.
// Fused epilogue: out = bev_query + conv + b2  -> query[0:QB]
// ---------------------------------------------------------------------------
__global__ __launch_bounds__(128) void conv5x5_kernel(
    const _Float16* __restrict__ X, const _Float16* __restrict__ Wr,
    const float* __restrict__ b2, const float* __restrict__ bevq,
    float* __restrict__ out)
{
  __shared__ __align__(16) _Float16 Wlds[2][128][128];   // 2 x 32KB
  const int h = blockIdx.y;
  const int w0 = blockIdx.x * 16;
  const int wave = threadIdx.x >> 5, lane = threadIdx.x & 31;
  const int l = lane & 15, hi = lane >> 4;

  v8f acc[2];
  for (int t = 0; t < 2; ++t) for (int r = 0; r < 8; ++r) acc[t][r] = 0.f;

  // each wave DMAs rows [wave*32, wave*32+32) of the tap slab (4096 halves)
  auto issue_tap = [&](int tap, int buf) {
    tdm_load_1d(Wr + ((size_t)tap * 128 + wave * 32) * 128,
                lds_off_of(&Wlds[buf][wave * 32][0]), 4096u);
  };

  issue_tap(0, 0);
  for (int tap = 0; tap < 25; ++tap) {
    int cur = tap & 1;
    if (tap + 1 < 25) {
      issue_tap(tap + 1, cur ^ 1);
      __builtin_amdgcn_s_wait_tensorcnt(1);   // slab `tap` landed (in-order)
    } else {
      __builtin_amdgcn_s_wait_tensorcnt(0);
    }
    __syncthreads();                          // all waves' chunks visible

    int dy = tap / 5 - 2, dx = tap % 5 - 2;
    int hh = h + dy;
    int wp = w0 + l + dx;                     // pixel for A row m = l
    bool valid = (hh >= 0) && (hh < 200) && (wp >= 0) && (wp < 200);
    const _Float16* src = X + (size_t)(hh * 200 + wp) * DCH;
    for (int kc = 0; kc < 128; kc += 32) {
      AFrag af;
      if (valid) {
        af.h[0] = *(const half8*)(src + kc + hi * 8);
        af.h[1] = *(const half8*)(src + kc + hi * 8 + 16);
      } else {
        for (int i = 0; i < 16; ++i) af.v[i] = (_Float16)0.f;
      }
      for (int t = 0; t < 2; ++t) {
        AFrag bf;
        int n0 = wave * 32 + t * 16;
        for (int i = 0; i < 16; ++i)
          bf.v[i] = Wlds[cur][kc + hi * 8 + (i < 8 ? i : i + 8)][n0 + l];
        acc[t] = __builtin_amdgcn_wmma_f32_16x16x32_f16(
            false, af.v, false, bf.v, (short)0, acc[t], false, false);
      }
    }
    __syncthreads();                          // done reading before overwrite
  }

  for (int t = 0; t < 2; ++t) {
    int n = wave * 32 + t * 16 + l;
    for (int r = 0; r < 8; ++r) {
      int m = r + hi * 8;
      int wq = w0 + m;
      if (wq < 200) {
        size_t q = (size_t)h * 200 + wq;
        out[q * DCH + n] = bevq[q * DCH + n] + acc[t][r] + b2[n];
      }
    }
  }
}

// tau[q][h] = det_qe[q] . tau_w[:,h] + tau_b[h]
__global__ void tau_kernel(const float* __restrict__ det_qe, const float* __restrict__ tw,
                           const float* __restrict__ tb, float* __restrict__ tau)
{
  int i = blockIdx.x * blockDim.x + threadIdx.x;
  if (i >= QD * 4) return;
  int q = i >> 2, h = i & 3;
  const float* x = det_qe + (size_t)q * DCH;
  float s = tb[h];
  for (int c = 0; c < DCH; ++c) s += x[c] * tw[c * 4 + h];
  tau[i] = s;
}

// ---------------------------------------------------------------------------
// Biased self-attention over 900 det queries. One wave per (q, head).
// ---------------------------------------------------------------------------
__global__ __launch_bounds__(256) void attn_kernel(
    const float* __restrict__ qh, const float* __restrict__ kh,
    const float* __restrict__ vh, const float* __restrict__ tau,
    const float* __restrict__ det_pos, float* __restrict__ sa)
{
  __shared__ float att_s[8][928];
  int wv = threadIdx.x >> 5, lane = threadIdx.x & 31;
  int gw = blockIdx.x * 8 + wv;
  if (gw >= QD * 4) return;
  int q = gw >> 2, h = gw & 3;
  const float* qrow = qh + (size_t)q * DCH + h * 32;
  float qr[32];
  for (int c = 0; c < 32; ++c) qr[c] = qrow[c];
  float cxq = det_pos[q * 3 + 0] * 100.f - 50.f;
  float cyq = det_pos[q * 3 + 1] * 100.f - 50.f;
  float tq = tau[q * 4 + h];
  const float scale = 0.17677669529663687f; // 1/sqrt(32)
  float sc[29];
  float mx = -1e30f;
  for (int i = 0; i < 29; ++i) {
    int k = lane + 32 * i;
    float s = -1e30f;
    if (k < QD) {
      const float* krow = kh + (size_t)k * DCH + h * 32;
      float d = 0.f;
      for (int c = 0; c < 32; ++c) d += qr[c] * krow[c];
      float dx = cxq - (det_pos[k * 3 + 0] * 100.f - 50.f);
      float dy = cyq - (det_pos[k * 3 + 1] * 100.f - 50.f);
      float dist = -sqrtf(dx * dx + dy * dy + 1e-12f);
      s = d * scale + tq * dist;
      mx = fmaxf(mx, s);
    }
    sc[i] = s;
  }
  for (int o = 16; o; o >>= 1) mx = fmaxf(mx, __shfl_xor(mx, o, 32));
  float sum = 0.f;
  for (int i = 0; i < 29; ++i) {
    int k = lane + 32 * i;
    float e = (k < QD) ? __expf(sc[i] - mx) : 0.f;
    sc[i] = e; sum += e;
  }
  for (int o = 16; o; o >>= 1) sum += __shfl_xor(sum, o, 32);
  float inv = 1.f / sum;
  for (int i = 0; i < 29; ++i) att_s[wv][lane + 32 * i] = sc[i] * inv;
  float acc = 0.f;
  for (int k = 0; k < QD; ++k) acc += att_s[wv][k] * vh[(size_t)k * DCH + h * 32 + lane];
  sa[(size_t)q * DCH + h * 32 + lane] = acc;
}

// LayerNorm over 128 channels; one wave per row.
__global__ __launch_bounds__(256) void ln_kernel(
    const float* __restrict__ X, const float* __restrict__ g,
    const float* __restrict__ b, float* __restrict__ Y, int M)
{
  int wv = threadIdx.x >> 5, lane = threadIdx.x & 31;
  int row = blockIdx.x * 8 + wv;
  if (row >= M) return;
  const float* x = X + (size_t)row * DCH;
  float a0 = x[lane], a1 = x[lane + 32], a2 = x[lane + 64], a3 = x[lane + 96];
  float s = a0 + a1 + a2 + a3;
  for (int o = 16; o; o >>= 1) s += __shfl_xor(s, o, 32);
  float mean = s * (1.f / 128.f);
  float d0 = a0 - mean, d1 = a1 - mean, d2 = a2 - mean, d3 = a3 - mean;
  float v = d0 * d0 + d1 * d1 + d2 * d2 + d3 * d3;
  for (int o = 16; o; o >>= 1) v += __shfl_xor(v, o, 32);
  float inv = rsqrtf(v * (1.f / 128.f) + 1e-5f);
  float* y = Y + (size_t)row * DCH;
  y[lane]      = d0 * inv * g[lane]      + b[lane];
  y[lane + 32] = d1 * inv * g[lane + 32] + b[lane + 32];
  y[lane + 64] = d2 * inv * g[lane + 64] + b[lane + 64];
  y[lane + 96] = d3 * inv * g[lane + 96] + b[lane + 96];
}

// ---------------------------------------------------------------------------
// Deformable sampling: one wave per (q,g,p), lane = channel within group.
// Output layout (Q, P, G, CG) = flat input of cmp MLP.
// ---------------------------------------------------------------------------
__global__ __launch_bounds__(256) void sample_kernel(
    const float* __restrict__ bev_pos, const float* __restrict__ det_pos,
    const float* __restrict__ l2i,
    const float* __restrict__ f0, const float* __restrict__ f1,
    const float* __restrict__ f2, const float* __restrict__ f3,
    const float* __restrict__ offb, const float* __restrict__ scb,
    float* __restrict__ sampled)
{
  int wv = threadIdx.x >> 5, lane = threadIdx.x & 31;
  long W = (long)blockIdx.x * 8 + wv;
  if (W >= (long)QT * 16) return;
  int q = (int)(W >> 4);
  int gp = (int)(W & 15);
  int g = gp >> 2, p = gp & 3;
  const float* pos = (q < QB) ? (bev_pos + (size_t)q * 3) : (det_pos + (size_t)(q - QB) * 3);
  const float* off = offb + (size_t)q * 48 + gp * 3;
  float px = pos[0] * 100.f - 50.f + off[0];
  float py = pos[1] * 100.f - 50.f + off[1];
  float pz = pos[2] * 8.f   - 5.f  + off[2];
  const float* sw = scb + (size_t)q * 64 + gp * 4;
  float s0 = sw[0], s1 = sw[1], s2 = sw[2], s3 = sw[3];
  float mx = fmaxf(fmaxf(s0, s1), fmaxf(s2, s3));
  float e0 = __expf(s0 - mx), e1 = __expf(s1 - mx), e2 = __expf(s2 - mx), e3 = __expf(s3 - mx);
  float isum = 1.f / (e0 + e1 + e2 + e3);
  float wl[4] = { e0 * isum, e1 * isum, e2 * isum, e3 * isum };
  const float* feats[4] = { f0, f1, f2, f3 };
  const int HH[4] = { 32, 16, 8, 4 };
  const int WW[4] = { 88, 44, 22, 11 };
  int c = g * 32 + lane;
  float acc = 0.f;
  for (int cam = 0; cam < NCAM; ++cam) {
    const float* Mc = l2i + cam * 16;
    float X = Mc[0] * px + Mc[1] * py + Mc[2]  * pz + Mc[3];
    float Y = Mc[4] * px + Mc[5] * py + Mc[6]  * pz + Mc[7];
    float Z = Mc[8] * px + Mc[9] * py + Mc[10] * pz + Mc[11];
    float valid = (Z > 1e-5f) ? 1.f : 0.f;
    float zs = fmaxf(Z, 1e-5f);
    float u = X / zs * (2.f / 704.f) - 1.f;
    float v = Y / zs * (2.f / 256.f) - 1.f;
    for (int lvl = 0; lvl < 4; ++lvl) {
      int Hh = HH[lvl], Ww = WW[lvl];
      float pxl = (u + 1.f) * 0.5f * (float)Ww - 0.5f;
      float pyl = (v + 1.f) * 0.5f * (float)Hh - 0.5f;
      float xf = floorf(pxl), yf = floorf(pyl);
      float wx = pxl - xf, wy = pyl - yf;
      int x0 = (int)xf, y0 = (int)yf;
      const float* fb = feats[lvl] + ((size_t)cam * DCH + c) * (size_t)(Hh * Ww);
      auto gather = [&](int xi, int yi) -> float {
        bool inb = (xi >= 0) && (xi < Ww) && (yi >= 0) && (yi < Hh);
        int xc = xi < 0 ? 0 : (xi > Ww - 1 ? Ww - 1 : xi);
        int yc = yi < 0 ? 0 : (yi > Hh - 1 ? Hh - 1 : yi);
        return inb ? fb[(size_t)yc * Ww + xc] : 0.f;
      };
      float v00 = gather(x0, y0),     v01 = gather(x0 + 1, y0);
      float v10 = gather(x0, y0 + 1), v11 = gather(x0 + 1, y0 + 1);
      float sval = (v00 * (1.f - wx) + v01 * wx) * (1.f - wy)
                 + (v10 * (1.f - wx) + v11 * wx) * wy;
      acc += sval * valid * wl[lvl];
    }
  }
  sampled[(((size_t)q * 4 + p) * 4 + g) * 32 + lane] = acc;
}

// ---------------------------------------------------------------------------
// Host side
// ---------------------------------------------------------------------------
static void run_gemm(int mode, const float* A, const float* B, const float* bias,
                     const float* Res, float* C, int M, int N, int K, hipStream_t s)
{
  dim3 grid((M + 127) / 128, (N + 63) / 64), blk(256);
  switch (mode) {
    case 0: gemm_f16_wmma<0,0,0><<<grid, blk, 0, s>>>(A, B, bias, Res, C, M, N, K); break; // plain
    case 1: gemm_f16_wmma<1,0,0><<<grid, blk, 0, s>>>(A, B, bias, Res, C, M, N, K); break; // relu
    case 2: gemm_f16_wmma<0,0,1><<<grid, blk, 0, s>>>(A, B, bias, Res, C, M, N, K); break; // +residual
    case 3: gemm_f16_wmma<2,1,0><<<grid, blk, 0, s>>>(A, B, bias, Res, C, M, N, K); break; // gelu, B^T
  }
}

extern "C" void kernel_launch(void* const* d_in, const int* in_sizes, int n_in,
                              void* d_out, int out_size, void* d_ws, size_t ws_size,
                              hipStream_t stream) {
  const float* bev_query = (const float*)d_in[0];
  const float* bev_pos   = (const float*)d_in[1];
  const float* det_query = (const float*)d_in[2];
  const float* det_pos   = (const float*)d_in[3];
  const float* lidar2img = (const float*)d_in[4];
  const float* feat0 = (const float*)d_in[5];
  const float* feat1 = (const float*)d_in[6];
  const float* feat2 = (const float*)d_in[7];
  const float* feat3 = (const float*)d_in[8];
  const float* pe_w1 = (const float*)d_in[9];
  const float* pe_b1 = (const float*)d_in[10];
  const float* pe_w2 = (const float*)d_in[11];
  const float* pe_b2 = (const float*)d_in[12];
  const float* inconv_w1 = (const float*)d_in[13];
  const float* inconv_b1 = (const float*)d_in[14];
  const float* inconv_w2 = (const float*)d_in[15];
  const float* inconv_b2 = (const float*)d_in[16];
  const float* tau_w = (const float*)d_in[17];
  const float* tau_b = (const float*)d_in[18];
  const float* wq = (const float*)d_in[19];
  const float* bq = (const float*)d_in[20];
  const float* wk = (const float*)d_in[21];
  const float* bk = (const float*)d_in[22];
  const float* wv = (const float*)d_in[23];
  const float* bv = (const float*)d_in[24];
  const float* wo = (const float*)d_in[25];
  const float* bo = (const float*)d_in[26];
  const float* off_w = (const float*)d_in[27];
  const float* off_b = (const float*)d_in[28];
  const float* sc_w = (const float*)d_in[29];
  const float* sc_b = (const float*)d_in[30];
  const float* cmp_w1 = (const float*)d_in[31];
  const float* cmp_b1 = (const float*)d_in[32];
  const float* cmp_w2 = (const float*)d_in[33];
  const float* cmp_b2 = (const float*)d_in[34];
  const float* cmp_w3 = (const float*)d_in[35];
  const float* cmp_b3 = (const float*)d_in[36];
  const float* ffn_w1 = (const float*)d_in[37];
  const float* ffn_b1 = (const float*)d_in[38];
  const float* ffn_w2 = (const float*)d_in[39];
  const float* ffn_b2 = (const float*)d_in[40];
  const float* ln1_g = (const float*)d_in[41];
  const float* ln1_b = (const float*)d_in[42];
  const float* ln2_g = (const float*)d_in[43];
  const float* ln2_b = (const float*)d_in[44];
  const float* ln3_g = (const float*)d_in[45];
  const float* ln3_b = (const float*)d_in[46];

  // Workspace carve (256B aligned), with lifetime-based aliasing.
  char* wsp = (char*)d_ws;
  auto carve = [&](size_t bytes) -> void* {
    void* pp = (void*)wsp;
    wsp += (bytes + 255) & ~(size_t)255;
    return pp;
  };
  float*     query   = (float*)carve((size_t)QT * DCH * 4);   // persistent activations
  float*     xin     = (float*)carve((size_t)QB * DCH * 4);   // bev_query + pe ; later off/sc
  float*     T1      = (float*)carve((size_t)QB * DCH * 4);   // gelu(conv1x1) f32
  _Float16*  A1h     = (_Float16*)carve((size_t)QB * DCH * 2);
  _Float16*  w2r     = (_Float16*)carve((size_t)25 * 128 * 128 * 2);
  float*     det_qe  = (float*)carve((size_t)QD * DCH * 4);
  float*     qhb     = (float*)carve((size_t)QD * DCH * 4);
  float*     khb     = (float*)carve((size_t)QD * DCH * 4);
  float*     vhb     = (float*)carve((size_t)QD * DCH * 4);
  float*     sab     = (float*)carve((size_t)QD * DCH * 4);
  float*     taub    = (float*)carve((size_t)QD * 4 * 4);
  float*     sampled = (float*)carve((size_t)QT * 512 * 4);
  float*     h1      = (float*)carve((size_t)QT * 512 * 4);
  float*     offb = xin;                        // xin dead after inconv1
  float*     scb  = xin + (size_t)QT * 48;
  float*     h2   = sampled;                    // sampled dead after cmp1
  float*     ffh  = h1;                         // h1 dead after cmp2/3

  // 1) positional-encoding MLPs (+ residual bases)
  pe_kernel<<<QT, 256, 0, stream>>>(bev_query, bev_pos, det_query, det_pos,
                                    pe_w1, pe_b1, pe_w2, pe_b2, xin, det_qe);
  // 2) 1x1 conv (GEMM, B^T, fused gelu)
  run_gemm(3, xin, inconv_w1, inconv_b1, nullptr, T1, QB, 128, 128, stream);
  // 3) f32 -> f16 activations for the implicit-GEMM conv
  cvt_f16_kernel<<<(QB * DCH + 255) / 256, 256, 0, stream>>>(T1, A1h, QB * DCH);
  // 4) weight swizzle (O,C,5,5) -> [tap][c][o] f16 (TDM source)
  w2prep_kernel<<<(25 * 128 * 128 + 255) / 256, 256, 0, stream>>>(inconv_w2, w2r);
  // 5) 5x5 conv (TDM-fed) + bias + residual -> query[0:QB]
  conv5x5_kernel<<<dim3(13, 200), 128, 0, stream>>>(A1h, w2r, inconv_b2, bev_query, query);
  // 6) det branch: tau, QKV projections, biased attention, out proj (+residual)
  tau_kernel<<<(QD * 4 + 127) / 128, 128, 0, stream>>>(det_qe, tau_w, tau_b, taub);
  run_gemm(0, det_qe,   wq, bq, nullptr, qhb, QD, 128, 128, stream);
  run_gemm(0, det_qe,   wk, bk, nullptr, khb, QD, 128, 128, stream);
  run_gemm(0, det_query, wv, bv, nullptr, vhb, QD, 128, 128, stream);
  attn_kernel<<<(QD * 4 + 7) / 8, 256, 0, stream>>>(qhb, khb, vhb, taub, det_pos, sab);
  run_gemm(2, sab, wo, bo, det_query, query + (size_t)QB * DCH, QD, 128, 128, stream);
  // 7) LN1 over the concatenated queries
  ln_kernel<<<(QT + 7) / 8, 256, 0, stream>>>(query, ln1_g, ln1_b, query, QT);
  // 8) offset / sampling-weight projections
  run_gemm(0, query, off_w, off_b, nullptr, offb, QT, 48, 128, stream);
  run_gemm(0, query, sc_w,  sc_b,  nullptr, scb,  QT, 64, 128, stream);
  // 9) deformable multi-cam multi-level sampling
  sample_kernel<<<(QT * 16 + 7) / 8, 256, 0, stream>>>(
      bev_pos, det_pos, lidar2img, feat0, feat1, feat2, feat3, offb, scb, sampled);
  // 10) cmp MLP (the 48-GFLOP GEMM chain) + residual
  run_gemm(1, sampled, cmp_w1, cmp_b1, nullptr, h1, QT, 512, 512, stream);
  run_gemm(1, h1,      cmp_w2, cmp_b2, nullptr, h2, QT, 512, 512, stream);
  run_gemm(2, h2,      cmp_w3, cmp_b3, query,  query, QT, 128, 512, stream);
  // 11) LN2, FFN (+residual), LN3 -> d_out (f32, concatenated bev|det)
  ln_kernel<<<(QT + 7) / 8, 256, 0, stream>>>(query, ln2_g, ln2_b, query, QT);
  run_gemm(1, query, ffn_w1, ffn_b1, nullptr, ffh, QT, 128, 128, stream);
  run_gemm(2, ffh,   ffn_w2, ffn_b2, query,  query, QT, 128, 128, stream);
  ln_kernel<<<(QT + 7) / 8, 256, 0, stream>>>(query, ln3_g, ln3_b, (float*)d_out, QT);
}
